// Model_541165879924
// MI455X (gfx1250) — matrix operand order repaired and checked by reading, not compiled, not run on hardware
//
#include <hip/hip_runtime.h>
#include <hip/hip_bf16.h>

typedef __attribute__((ext_vector_type(16))) _Float16 v16h;
typedef __attribute__((ext_vector_type(8)))  float    v8f;

#if defined(__has_builtin)
#if __has_builtin(__builtin_amdgcn_tensor_load_to_lds)
#define HAVE_TDM 1
#endif
#endif

// ---------------- constants ----------------
#define NTOK 4096           // 64 images * 8*8 tokens
#define CH   128
#define VQ_ROWS 128         // codes staged per LDS chunk in argmin
__constant__ int c_DLIST[8] = {128, 64, 32, 16, 8, 4, 2, 1};

// output layout (f32): x_recon | loss | recon | vqloss[8] | grad_h | h
#define OFF_XR    0
#define OFF_LOSS  196608
#define OFF_RECON 196609
#define OFF_VQL   196610
#define OFF_GH    196618
#define OFF_H     720906

// GEMM flags
#define GF_RELU   1
#define GF_RESID  2
#define GF_F16OUT 4
#define GF_BIAS   8
// conv flags
#define CF_RELU   1
#define CF_OUTTOK 2
#define CF_BIAS   4
#define CF_INTOK  8

// ---------------- WMMA fragment loaders (contiguous k = 16*(lane/16) + s, same for A and B) ----------------
// A fragment, no bounds check (full tile)
__device__ __forceinline__ v16h load_a_frag_f(const _Float16* __restrict__ A, int lda,
                                              int m0, int k0, int lane) {
  int row = m0 + (lane & 15);
  int hi  = lane >> 4;
  const _Float16* p = A + (size_t)row * lda + k0 + hi * 16;
  v16h a;
#pragma unroll
  for (int h = 0; h < 16; ++h) {
    int g = h >> 1;
    int k = h + (g & 0); // lanes0-15: K 0-15 ; +16 for hi half
    a[h] = p[k];
  }
  return a;
}
// A fragment, branchless row guard (clamp + select-zero; no exec divergence)
__device__ __forceinline__ v16h load_a_frag_g(const _Float16* __restrict__ A, int lda, int M,
                                              int m0, int k0, int lane) {
  int row = m0 + (lane & 15);
  bool ok = row < M;
  int r = ok ? row : (M - 1);
  const _Float16* p = A + (size_t)r * lda + k0 + (lane >> 4) * 16;
  v16h a;
#pragma unroll
  for (int h = 0; h < 16; ++h) {
    int g = h >> 1;
    int k = h + (g & 0);
    _Float16 v = p[k];
    a[h] = ok ? v : (_Float16)0.f;
  }
  return a;
}
// B fragment from W[N, ldw] row-major (B[k][n] = W[n][k]), full tile
__device__ __forceinline__ v16h load_b_frag_f(const _Float16* __restrict__ W, int ldw,
                                              int n0, int k0, int lane) {
  int col = n0 + (lane & 15);
  const _Float16* p = W + (size_t)col * ldw + k0 + (lane >> 4) * 16;
  v16h b;
#pragma unroll
  for (int h = 0; h < 16; ++h) b[h] = p[h];
  return b;
}
// B fragment, branchless col guard
__device__ __forceinline__ v16h load_b_frag_g(const _Float16* __restrict__ W, int ldw, int N,
                                              int n0, int k0, int lane) {
  int col = n0 + (lane & 15);
  bool ok = col < N;
  int c = ok ? col : (N - 1);
  const _Float16* p = W + (size_t)c * ldw + k0 + (lane >> 4) * 16;
  v16h b;
#pragma unroll
  for (int h = 0; h < 16; ++h) {
    _Float16 v = p[h];
    b[h] = ok ? v : (_Float16)0.f;
  }
  return b;
}

__device__ __forceinline__ float wave_reduce_sum(float v) {
#pragma unroll
  for (int o = 16; o > 0; o >>= 1) v += __shfl_down(v, o, 32);
  return v;
}

// C tile epilogue (16x16), handles bias/resid/relu/f16 copy + ragged edges
__device__ __forceinline__ void store_tile(v8f acc, int m0, int n0,
                                           const float* __restrict__ bias,
                                           const float* __restrict__ resid,
                                           float* __restrict__ C, _Float16* __restrict__ C16,
                                           int M, int N, int flags, int lane) {
  int col = lane & 15, hi = lane >> 4;
  int n = n0 + col;
  if (n >= N) return;
  float bv = (flags & GF_BIAS) ? bias[n] : 0.f;
#pragma unroll
  for (int r = 0; r < 8; ++r) {
    int m = m0 + r + hi * 8;
    if (m >= M) continue;
    float v = acc[r] + bv;
    if (flags & GF_RESID) v += resid[(size_t)m * N + n];
    if (flags & GF_RELU)  v = fmaxf(v, 0.f);
    C[(size_t)m * N + n] = v;
    if (flags & GF_F16OUT) C16[(size_t)m * N + n] = (_Float16)v;
  }
}

// ---------------- generic WMMA GEMM: C[M,N] = A[M,K] * W[N,K]^T ----------------
// Each wave owns a 16x(16*gsz) strip: A fragment reused across gsz back-to-back WMMAs.
__global__ void k_wmma_gemm(const _Float16* __restrict__ A, const _Float16* __restrict__ W,
                            const float* __restrict__ bias, const float* __restrict__ resid,
                            float* __restrict__ C, _Float16* __restrict__ C16,
                            int M, int N, int K, int flags) {
  int lane = threadIdx.x & 31;
  int gw   = (blockIdx.x * blockDim.x + threadIdx.x) >> 5;
  int mt = (M + 15) >> 4, nt = (N + 15) >> 4;
  int gsz = ((nt & 3) == 0) ? 4 : (((nt & 1) == 0) ? 2 : 1);
  int ng  = (gsz == 4) ? (nt >> 2) : ((gsz == 2) ? (nt >> 1) : nt);
  if (gw >= mt * ng) return;
  int m0 = (gw / ng) << 4;
  int n0 = (gw % ng) * (gsz << 4);
  bool full = (m0 + 16 <= M) && (n0 + (gsz << 4) <= N);

  if (full && gsz == 4) {
    v8f a0 = {}, a1 = {}, a2 = {}, a3 = {};
    for (int k0 = 0; k0 < K; k0 += 32) {
      if (k0 + 32 < K) {
        __builtin_prefetch(A + (size_t)(m0 + (lane & 15)) * K + k0 + 32, 0, 1);
        __builtin_prefetch(W + (size_t)(n0 + (lane & 15)) * K + k0 + 32, 0, 1);
      }
      v16h a  = load_a_frag_f(A, K, m0, k0, lane);
      v16h b0 = load_b_frag_f(W, K, n0,      k0, lane);
      v16h b1 = load_b_frag_f(W, K, n0 + 16, k0, lane);
      v16h b2 = load_b_frag_f(W, K, n0 + 32, k0, lane);
      v16h b3 = load_b_frag_f(W, K, n0 + 48, k0, lane);
      a0 = __builtin_amdgcn_wmma_f32_16x16x32_f16(false, a, false, b0, (short)0, a0, false, false);
      a1 = __builtin_amdgcn_wmma_f32_16x16x32_f16(false, a, false, b1, (short)0, a1, false, false);
      a2 = __builtin_amdgcn_wmma_f32_16x16x32_f16(false, a, false, b2, (short)0, a2, false, false);
      a3 = __builtin_amdgcn_wmma_f32_16x16x32_f16(false, a, false, b3, (short)0, a3, false, false);
    }
    store_tile(a0, m0, n0,      bias, resid, C, C16, M, N, flags, lane);
    store_tile(a1, m0, n0 + 16, bias, resid, C, C16, M, N, flags, lane);
    store_tile(a2, m0, n0 + 32, bias, resid, C, C16, M, N, flags, lane);
    store_tile(a3, m0, n0 + 48, bias, resid, C, C16, M, N, flags, lane);
  } else if (full && gsz == 2) {
    v8f a0 = {}, a1 = {};
    for (int k0 = 0; k0 < K; k0 += 32) {
      if (k0 + 32 < K) __builtin_prefetch(A + (size_t)(m0 + (lane & 15)) * K + k0 + 32, 0, 1);
      v16h a  = load_a_frag_f(A, K, m0, k0, lane);
      v16h b0 = load_b_frag_f(W, K, n0,      k0, lane);
      v16h b1 = load_b_frag_f(W, K, n0 + 16, k0, lane);
      a0 = __builtin_amdgcn_wmma_f32_16x16x32_f16(false, a, false, b0, (short)0, a0, false, false);
      a1 = __builtin_amdgcn_wmma_f32_16x16x32_f16(false, a, false, b1, (short)0, a1, false, false);
    }
    store_tile(a0, m0, n0,      bias, resid, C, C16, M, N, flags, lane);
    store_tile(a1, m0, n0 + 16, bias, resid, C, C16, M, N, flags, lane);
  } else {
    for (int t = 0; t < gsz; ++t) {
      int nb = n0 + (t << 4);
      v8f acc = {};
      for (int k0 = 0; k0 < K; k0 += 32) {
        v16h a = load_a_frag_g(A, K, M, m0, k0, lane);
        v16h b = load_b_frag_g(W, K, N, nb, k0, lane);
        acc = __builtin_amdgcn_wmma_f32_16x16x32_f16(false, a, false, b, (short)0, acc, false, false);
      }
      store_tile(acc, m0, nb, bias, resid, C, C16, M, N, flags, lane);
    }
  }
}

// ---------------- TDM staging helper (async tensor load, §8.3/8.4 D# packing) ----------------
#ifdef HAVE_TDM
typedef __attribute__((ext_vector_type(4))) unsigned int u32x4_t;
typedef __attribute__((ext_vector_type(8))) int          i32x8_t;
typedef __attribute__((ext_vector_type(4))) int          i32x4_t;

__device__ __forceinline__ void tdm_stage_rows(const _Float16* gptr, unsigned lds_off,
                                               int elemsPerRow, int rows) {
  unsigned long long ga = (unsigned long long)gptr;
  u32x4_t g0;
  g0[0] = 1u;                                             // count=1, user descriptor
  g0[1] = lds_off;                                        // LDS byte address (dyn-LDS base)
  g0[2] = (unsigned)(ga & 0xFFFFFFFFull);                 // global_addr[31:0]
  g0[3] = (unsigned)((ga >> 32) & 0x01FFFFFFull) | (2u << 30); // addr[56:32] | type=2 (image)
  unsigned e = (unsigned)elemsPerRow, r = (unsigned)rows;
  i32x8_t g1;
  g1[0] = (int)(1u << 16);                                // data_size=1 -> 2-byte elements
  g1[1] = (int)((e & 0xFFFFu) << 16);                     // tensor_dim0[15:0] @ bits63:48
  g1[2] = (int)((e >> 16) | ((r & 0xFFFFu) << 16));       // dim0[31:16] | tensor_dim1[15:0]
  g1[3] = (int)((r >> 16) | ((e & 0xFFFFu) << 16));       // dim1[31:16] | tile_dim0
  g1[4] = (int)(r & 0xFFFFu);                             // tile_dim1 (tile_dim2 = 0)
  g1[5] = (int)e;                                         // tensor_dim0_stride[31:0]
  g1[6] = 0;                                              // stride0[47:32] | stride1[15:0]
  g1[7] = 0;
  i32x4_t z4 = {0, 0, 0, 0};
#if __clang_major__ >= 23
  i32x8_t z8 = {0, 0, 0, 0, 0, 0, 0, 0};
  __builtin_amdgcn_tensor_load_to_lds(g0, g1, z4, z4, z8, 0);
#else
  __builtin_amdgcn_tensor_load_to_lds(g0, g1, z4, z4, 0);
#endif
}
#endif

// ---------------- VQ: fused distance-GEMM + argmin, codebook streamed through LDS ------------
// grid: (M/16, chunks); block: 256 (8 waves). Each block stages a 128-code chunk of E into LDS
// (TDM when available), then each wave runs WMMA on one 16-code tile from LDS.
__global__ void k_vq_argmin(const _Float16* __restrict__ Z, const _Float16* __restrict__ E,
                            const float* __restrict__ enorm, unsigned long long* __restrict__ best,
                            int M, int Kcode, int dpad) {
  extern __shared__ _Float16 sE[];   // [VQ_ROWS][dpad]
  int lane = threadIdx.x & 31;
  int wv   = threadIdx.x >> 5;
  int m0   = blockIdx.x << 4;
  int nk   = dpad >> 5;
  v16h afr[4];
#pragma unroll
  for (int i = 0; i < 4; ++i)
    if (i < nk) afr[i] = load_a_frag_f(Z, dpad, m0, i * 32, lane);

  for (int c0 = blockIdx.y * VQ_ROWS; c0 < Kcode; c0 += gridDim.y * VQ_ROWS) {
#ifdef HAVE_TDM
    if (wv == 0)
      tdm_stage_rows(E + (size_t)c0 * dpad, 0u, dpad, VQ_ROWS);
    __builtin_amdgcn_s_wait_tensorcnt(0);
    __syncthreads();
#else
    {
      unsigned long long* d8 = (unsigned long long*)sE;
      const unsigned long long* s8 = (const unsigned long long*)(E + (size_t)c0 * dpad);
      int n8 = (VQ_ROWS * dpad) >> 2;
      for (int i = threadIdx.x; i < n8; i += blockDim.x) d8[i] = s8[i];
    }
    __syncthreads();
#endif
    int n0l = wv << 4;                               // one 16-code tile per wave
    v8f acc = {};
#pragma unroll
    for (int i = 0; i < 4; ++i) {
      if (i >= nk) break;
      v16h b = load_b_frag_f(sE, dpad, n0l, i * 32, lane);
      acc = __builtin_amdgcn_wmma_f32_16x16x32_f16(false, afr[i], false, b, (short)0, acc, false, false);
    }
    int col = lane & 15, hi = lane >> 4;
    int n = c0 + n0l + col;
    float en = enorm[n];
#pragma unroll
    for (int r = 0; r < 8; ++r) {
      float v = en - 2.0f * acc[r];                  // ||z||^2 constant per row -> dropped
      unsigned ub = __float_as_uint(v);
      ub = (v < 0.f) ? ~ub : (ub | 0x80000000u);     // order-preserving float->uint
      unsigned long long key = ((unsigned long long)ub << 32) | (unsigned)n;
#pragma unroll
      for (int d = 8; d >= 1; d >>= 1) {
        unsigned long long o = __shfl_xor(key, d, 32);
        if (o < key) key = o;
      }
      if (col == 0) atomicMin(&best[m0 + r + hi * 8], key);
    }
    __syncthreads();                                 // protect LDS before next chunk
  }
}

__global__ void k_vq_finish(const unsigned long long* __restrict__ best, const float* __restrict__ E,
                            const float* __restrict__ Zf, float* __restrict__ Q,
                            float* __restrict__ accum, int M, int d) {
  int i = blockIdx.x * blockDim.x + threadIdx.x;
  float sq = 0.f;
  if (i < M * d) {
    int m = i / d, j = i % d;
    int idx = (int)(best[m] & 0xFFFFFFFFull);
    float e = E[(size_t)idx * d + j];
    float z = Zf[i];
    Q[i] = e;
    float df = z - e;
    sq = df * df;
  }
  sq = wave_reduce_sum(sq);
  if ((threadIdx.x & 31) == 0) atomicAdd(accum, sq);
}

__global__ void k_enorm(const float* __restrict__ E, float* __restrict__ nrm, int K, int d) {
  int k = blockIdx.x * blockDim.x + threadIdx.x;
  if (k >= K) return;
  const float* p = E + (size_t)k * d;
  float s = 0.f;
  for (int j = 0; j < d; ++j) s += p[j] * p[j];
  nrm[k] = s;
}

// ---------------- layout / conversion helpers ----------------
__global__ void k_cvt_f16(const float* __restrict__ s, _Float16* __restrict__ d, int n) {
  int i = blockIdx.x * blockDim.x + threadIdx.x;
  if (i < n) d[i] = (_Float16)s[i];
}
__global__ void k_pad_rows_f16(const float* __restrict__ s, _Float16* __restrict__ d,
                               int R, int C, int Cp) {
  int i = blockIdx.x * blockDim.x + threadIdx.x;
  if (i >= R * Cp) return;
  int c = i % Cp, r = i / Cp;
  d[i] = (c < C) ? (_Float16)s[(size_t)r * C + c] : (_Float16)0.f;
}
__global__ void k_transpose_pad_f16(const float* __restrict__ s, _Float16* __restrict__ d,
                                    int R, int C, int Rp) {
  int i = blockIdx.x * blockDim.x + threadIdx.x;
  if (i >= C * Rp) return;
  int r = i % Rp, c = i / Rp;
  d[i] = (r < R) ? (_Float16)s[(size_t)r * C + c] : (_Float16)0.f;
}
__global__ void k_repack3x3_rot(const float* __restrict__ w, _Float16* __restrict__ d,
                                int Co, int Ci, int Kp) {
  int i = blockIdx.x * blockDim.x + threadIdx.x;
  if (i >= Ci * Kp) return;
  int kc = i % Kp, ci = i / Kp;
  _Float16 v = (_Float16)0.f;
  if (kc < Co * 9) {
    int co = kc / 9, t = kc % 9;
    v = (_Float16)w[((size_t)(co * Ci + ci)) * 9 + (8 - t)];
  }
  d[i] = v;
}
__global__ void k_im2col3x3_tok(const float* __restrict__ T, _Float16* __restrict__ col,
                                int C, int Kp) {
  int i = blockIdx.x * blockDim.x + threadIdx.x;
  if (i >= NTOK * Kp) return;
  int kc = i % Kp, row = i / Kp;
  _Float16 v = (_Float16)0.f;
  if (kc < C * 9) {
    int c = kc / 9, t = kc % 9, ky = t / 3, kx = t % 3;
    int x = row & 7, y = (row >> 3) & 7, b = row >> 6;
    int iy = y + ky - 1, ix = x + kx - 1;
    if ((unsigned)iy < 8u && (unsigned)ix < 8u)
      v = (_Float16)T[((size_t)(b * 64) + iy * 8 + ix) * C + c];
  }
  col[i] = v;
}
__global__ void k_tok_to_nchw(const float* __restrict__ T, float* __restrict__ out,
                              int B, int C, int HW) {
  int i = blockIdx.x * blockDim.x + threadIdx.x;
  if (i >= B * C * HW) return;
  int s = i % HW, c = (i / HW) % C, b = i / (HW * C);
  out[i] = T[((size_t)b * HW + s) * C + c];
}

// ---------------- direct convs (strided 4x4, stride2 pad1) ----------------
__global__ void k_conv4x4s2(const float* __restrict__ in, const float* __restrict__ w,
                            const float* __restrict__ bias, float* __restrict__ out,
                            int B, int Ci, int Co, int Hi, int Wi, int flags) {
  int Ho = Hi >> 1, Wo = Wi >> 1;
  int idx = blockIdx.x * blockDim.x + threadIdx.x;
  if (idx >= B * Co * Ho * Wo) return;
  int x = idx % Wo, y = (idx / Wo) % Ho, co = (idx / (Wo * Ho)) % Co, b = idx / (Wo * Ho * Co);
  float acc = (flags & CF_BIAS) ? bias[co] : 0.f;
  for (int ci = 0; ci < Ci; ++ci) {
    const float* ip = in + ((size_t)(b * Ci + ci)) * Hi * Wi;
    const float* wp = w + ((size_t)(co * Ci + ci)) * 16;
#pragma unroll
    for (int ky = 0; ky < 4; ++ky) {
      int iy = y * 2 - 1 + ky;
      if ((unsigned)iy >= (unsigned)Hi) continue;
#pragma unroll
      for (int kx = 0; kx < 4; ++kx) {
        int ix = x * 2 - 1 + kx;
        if ((unsigned)ix >= (unsigned)Wi) continue;
        acc += ip[iy * Wi + ix] * wp[ky * 4 + kx];
      }
    }
  }
  if (flags & CF_RELU) acc = fmaxf(acc, 0.f);
  if (flags & CF_OUTTOK) out[((size_t)(b * Ho * Wo) + y * Wo + x) * Co + co] = acc;
  else                   out[idx] = acc;
}
__global__ void k_convT4x4s2(const float* __restrict__ in, const float* __restrict__ w,
                             const float* __restrict__ bias, float* __restrict__ out,
                             int B, int Ci, int Co, int Hi, int Wi, int flags) {
  int Ho = Hi << 1, Wo = Wi << 1;
  int idx = blockIdx.x * blockDim.x + threadIdx.x;
  if (idx >= B * Co * Ho * Wo) return;
  int x = idx % Wo, y = (idx / Wo) % Ho, co = (idx / (Wo * Ho)) % Co, b = idx / (Wo * Ho * Co);
  float acc = bias ? bias[co] : 0.f;
  for (int ci = 0; ci < Ci; ++ci) {
#pragma unroll
    for (int ky = 0; ky < 4; ++ky) {
      int ty = y + 1 - ky;
      if (ty & 1) continue;
      int iy = ty >> 1;
      if ((unsigned)iy >= (unsigned)Hi) continue;
#pragma unroll
      for (int kx = 0; kx < 4; ++kx) {
        int tx = x + 1 - kx;
        if (tx & 1) continue;
        int ix = tx >> 1;
        if ((unsigned)ix >= (unsigned)Wi) continue;
        float iv = (flags & CF_INTOK)
                     ? in[((size_t)(b * Hi * Wi) + iy * Wi + ix) * Ci + ci]
                     : in[((size_t)(b * Ci + ci) * Hi + iy) * Wi + ix];
        acc += iv * w[((size_t)(ci * Co + co)) * 16 + ky * 4 + kx];
      }
    }
  }
  if (flags & CF_RELU) acc = fmaxf(acc, 0.f);
  out[idx] = acc;
}

// ---------------- attention (per token; keys staged in LDS) ----------------
__global__ void k_attn_fwd(const float* __restrict__ q, const float* __restrict__ kk,
                           const float* __restrict__ u, int* __restrict__ sel,
                           float* __restrict__ ysoft, float* __restrict__ phead, int M) {
  __shared__ float skk[1024];
  for (int i = threadIdx.x; i < 1024; i += blockDim.x) skk[i] = kk[i];
  __syncthreads();
  int m = blockIdx.x * blockDim.x + threadIdx.x;
  if (m >= M) return;
  const float* qp = q + (size_t)m * 128;
  float p[2][8];
  for (int h = 0; h < 2; ++h) {
    float s[8]; float mx = -1e30f;
    for (int j = 0; j < 8; ++j) {
      float a = 0.f;
      for (int d = 0; d < 64; ++d) a += qp[h * 64 + d] * skk[j * 128 + h * 64 + d];
      s[j] = a * 0.125f;
      mx = fmaxf(mx, s[j]);
    }
    float sm = 0.f;
    for (int j = 0; j < 8; ++j) { s[j] = __expf(s[j] - mx); sm += s[j]; }
    float inv = 1.f / sm;
    for (int j = 0; j < 8; ++j) { p[h][j] = s[j] * inv; phead[(size_t)m * 16 + h * 8 + j] = p[h][j]; }
  }
  float l[8]; float mx = -1e30f;
  for (int j = 0; j < 8; ++j) {
    float pm = 0.5f * (p[0][j] + p[1][j]);
    float uu = u[(size_t)m * 8 + j];
    float g = -__logf(-__logf(uu + 1e-10f) + 1e-10f);
    l[j] = pm + g;
    mx = fmaxf(mx, l[j]);
  }
  float sm = 0.f;
  for (int j = 0; j < 8; ++j) { l[j] = __expf(l[j] - mx); sm += l[j]; }
  float inv = 1.f / sm; int am = 0; float bv = -1.f;
  for (int j = 0; j < 8; ++j) {
    float ys = l[j] * inv;
    ysoft[(size_t)m * 8 + j] = ys;
    if (ys > bv) { bv = ys; am = j; }
  }
  sel[m] = am;
}

__global__ void k_att_bwd(const float* __restrict__ gstate, const float* __restrict__ nv,
                          const float* __restrict__ kk, const float* __restrict__ ysoft,
                          const float* __restrict__ phead, float* __restrict__ dq, int M) {
  __shared__ float skk[1024];
  for (int i = threadIdx.x; i < 1024; i += blockDim.x) skk[i] = kk[i];
  __syncthreads();
  int m = blockIdx.x * blockDim.x + threadIdx.x;
  if (m >= M) return;
  const float* gp = gstate + (size_t)m * 128;
  float gj[8];
  for (int j = 0; j < 8; ++j) {
    const float* zp = nv + (size_t)j * (NTOK * CH) + (size_t)m * 128;
    float a = 0.f;
    for (int c = 0; c < 128; ++c) a += gp[c] * zp[c];
    gj[j] = a;
  }
  const float* ys = ysoft + (size_t)m * 8;
  float dot = 0.f;
  for (int j = 0; j < 8; ++j) dot += gj[j] * ys[j];
  float dpm[8];
  for (int j = 0; j < 8; ++j) dpm[j] = ys[j] * (gj[j] - dot);
  for (int h = 0; h < 2; ++h) {
    const float* ph = phead + (size_t)m * 16 + h * 8;
    float dd = 0.f;
    for (int j = 0; j < 8; ++j) dd += 0.5f * dpm[j] * ph[j];
    float ds[8];
    for (int j = 0; j < 8; ++j) ds[j] = ph[j] * (0.5f * dpm[j] - dd);
    for (int d = 0; d < 64; ++d) {
      float a = 0.f;
      for (int j = 0; j < 8; ++j) a += ds[j] * skk[j * 128 + h * 64 + d];
      dq[(size_t)m * 128 + h * 64 + d] = 0.125f * a;
    }
  }
}

// ---------------- elementwise / reductions ----------------
__global__ void k_relu(const float* __restrict__ a, float* __restrict__ b, int n) {
  int i = blockIdx.x * blockDim.x + threadIdx.x;
  if (i < n) b[i] = fmaxf(a[i], 0.f);
}
__global__ void k_relu_bwd(float* __restrict__ g, const float* __restrict__ act, int n) {
  int i = blockIdx.x * blockDim.x + threadIdx.x;
  if (i < n) g[i] = (act[i] > 0.f) ? g[i] : 0.f;
}
__global__ void k_add(float* __restrict__ a, const float* __restrict__ b, int n) {
  int i = blockIdx.x * blockDim.x + threadIdx.x;
  if (i < n) a[i] += b[i];
}
__global__ void k_mse_grad(const float* __restrict__ x, const float* __restrict__ xr,
                           float* __restrict__ g, int n, float scale) {
  int i = blockIdx.x * blockDim.x + threadIdx.x;
  if (i < n) g[i] = scale * (xr[i] - x[i]);
}
__global__ void k_vq_grad_z(float* __restrict__ dz, const float* __restrict__ z,
                            const float* __restrict__ q, int n, float coef) {
  int i = blockIdx.x * blockDim.x + threadIdx.x;
  if (i < n) dz[i] = dz[i] + coef * (z[i] - q[i]);
}
__global__ void k_select_grad(const int* __restrict__ sel, const float* __restrict__ g,
                              float* __restrict__ gz, int lvl, int M) {
  int i = blockIdx.x * blockDim.x + threadIdx.x;
  if (i >= M * CH) return;
  gz[i] = (sel[i / CH] == lvl) ? g[i] : 0.f;
}
__global__ void k_combine_state(const int* __restrict__ sel, const float* __restrict__ nv,
                                float* __restrict__ st, int M) {
  int i = blockIdx.x * blockDim.x + threadIdx.x;
  if (i >= M * CH) return;
  st[i] = nv[(size_t)sel[i / CH] * (NTOK * CH) + i];
}
__global__ void k_sqdiff(const float* __restrict__ a, const float* __restrict__ b,
                         float* __restrict__ accum, int n) {
  int i = blockIdx.x * blockDim.x + threadIdx.x;
  float s = 0.f;
  if (i < n) { float d = a[i] - b[i]; s = d * d; }
  s = wave_reduce_sum(s);
  if ((threadIdx.x & 31) == 0) atomicAdd(accum, s);
}
__global__ void k_finalize(const float* __restrict__ accum, float* __restrict__ out) {
  float recon = accum[0] / 196608.f;
  float loss = recon;
  out[OFF_RECON] = recon;
  for (int i = 0; i < 8; ++i) {
    float v = 1.25f * accum[1 + i] / (4096.f * (float)c_DLIST[i]);
    out[OFF_VQL + i] = v;
    loss += v * 0.125f;
  }
  out[OFF_LOSS] = loss;
}

// =============================== host orchestration ===============================
struct Bump {
  char* base; size_t off; size_t cap;
  void* take(size_t bytes) {
    size_t a = (off + 255) & ~(size_t)255;
    off = a + bytes;
    return base + a;
  }
  float*              f32(size_t n) { return (float*)take(n * 4); }
  _Float16*           f16(size_t n) { return (_Float16*)take(n * 2); }
  int*                i32(size_t n) { return (int*)take(n * 4); }
  unsigned long long* u64(size_t n) { return (unsigned long long*)take(n * 8); }
};

static inline int cdiv(int a, int b) { return (a + b - 1) / b; }

extern "C" void kernel_launch(void* const* d_in, const int* in_sizes, int n_in,
                              void* d_out, int out_size, void* d_ws, size_t ws_size,
                              hipStream_t stream) {
  const int KL[8]   = {512, 1024, 2048, 4096, 8192, 16384, 32768, 65536};
  const int DL[8]   = {128, 64, 32, 16, 8, 4, 2, 1};
  int DP[8], K9P[8];
  for (int i = 0; i < 8; ++i) {
    DP[i]  = DL[i] < 32 ? 32 : DL[i];
    K9P[i] = ((DL[i] * 9) + 31) & ~31;
  }
  const float* in_f[70];
  for (int i = 0; i < n_in && i < 70; ++i) in_f[i] = (const float*)d_in[i];
  const float* x   = in_f[0];
  const float* gum = in_f[1];
  const float* qk  = in_f[2];
  const float* wq  = in_f[3]; const float* bq = in_f[4];
  const float* wk  = in_f[5]; const float* bk = in_f[6];
  const float* c1w = in_f[7];  const float* c1b = in_f[8];
  const float* c2w = in_f[9];  const float* c2b = in_f[10];
  const float* c3w = in_f[11]; const float* c3b = in_f[12];
  const float* ewa[2] = {in_f[13], in_f[15]};
  const float* ewb[2] = {in_f[14], in_f[16]};
  const float *mcw[8], *mcb[8], *emb[8], *m1w[8], *m1b[8];
  for (int i = 0; i < 8; ++i) {
    mcw[i] = in_f[17 + i]; mcb[i] = in_f[25 + i];
    emb[i] = in_f[33 + i]; m1w[i] = in_f[41 + i]; m1b[i] = in_f[49 + i];
  }
  const float* dwa[2] = {in_f[57], in_f[59]};
  const float* dwb[2] = {in_f[58], in_f[60]};
  const float* t1w = in_f[61]; const float* t1b = in_f[62];
  const float* t2w = in_f[63]; const float* t2b = in_f[64];
  float* out = (float*)d_out;

  Bump ws{(char*)d_ws, 0, ws_size};
  _Float16* w_c3_16 = ws.f16(128 * 1152);
  _Float16 *ewa16[2], *ewb16[2], *dwa16[2], *dwb16[2], *dwbT16[2], *dwarot16[2];
  for (int b = 0; b < 2; ++b) {
    ewa16[b] = ws.f16(32 * 1152); ewb16[b] = ws.f16(128 * 32);
    dwa16[b] = ws.f16(32 * 1152); dwb16[b] = ws.f16(128 * 32);
    dwbT16[b] = ws.f16(32 * 128); dwarot16[b] = ws.f16(128 * 288);
  }
  _Float16* wq16  = ws.f16(128 * 128);
  _Float16* wk16  = ws.f16(128 * 128);
  _Float16* wqT16 = ws.f16(128 * 128);
  _Float16* qk16  = ws.f16(8 * 128);
  _Float16 *mcw16[8], *mcwT16[8], *m1w16[8], *m1rot16[8], *E16[8], *Z16[8];
  float *enorm[8], *Zf[8], *Qb[8];
  for (int i = 0; i < 8; ++i) {
    mcw16[i]  = ws.f16((size_t)DL[i] * 128);
    mcwT16[i] = ws.f16((size_t)128 * DP[i]);
    m1w16[i]  = ws.f16((size_t)128 * K9P[i]);
    m1rot16[i] = ws.f16((size_t)DL[i] * 1152);
    E16[i]    = ws.f16((size_t)KL[i] * DP[i]);
    Z16[i]    = ws.f16((size_t)NTOK * DP[i]);
    enorm[i]  = ws.f32(KL[i]);
    Zf[i]     = ws.f32((size_t)NTOK * DL[i]);
    Qb[i]     = ws.f32((size_t)NTOK * DL[i]);
  }
  _Float16* col  = ws.f16((size_t)NTOK * 1152);
  _Float16* t16  = ws.f16((size_t)NTOK * 32);
  _Float16* h16  = ws.f16((size_t)NTOK * 128);
  _Float16* g16  = ws.f16((size_t)NTOK * 128);
  _Float16* dq16 = ws.f16((size_t)NTOK * 128);
  _Float16* dzp16 = ws.f16((size_t)NTOK * 128);
  float* a1    = ws.f32(64 * 64 * 16 * 16);
  float* a2tok = ws.f32((size_t)NTOK * 128);
  float* x0    = ws.f32((size_t)NTOK * 128);
  float* ubuf  = ws.f32((size_t)NTOK * 128);
  float* tbuf  = ws.f32((size_t)NTOK * 32);
  float* x1    = ws.f32((size_t)NTOK * 128);
  float* x2    = ws.f32((size_t)NTOK * 128);
  float* htok  = ws.f32((size_t)NTOK * 128);
  float* q32   = ws.f32((size_t)NTOK * 128);
  float* kk32  = ws.f32(16 * 128);
  float* nv    = ws.f32((size_t)8 * NTOK * 128);
  float* state = ws.f32((size_t)NTOK * 128);
  float* u0d = ws.f32((size_t)NTOK * 128); float* t0d = ws.f32((size_t)NTOK * 32);
  float* u1d = ws.f32((size_t)NTOK * 128); float* t1d = ws.f32((size_t)NTOK * 32);
  float* ds1 = ws.f32((size_t)NTOK * 128); float* ds2 = ws.f32((size_t)NTOK * 128);
  float* ytok = ws.f32((size_t)NTOK * 128);
  float* d1   = ws.f32(64 * 64 * 16 * 16);
  float* gs   = ws.f32((size_t)NTOK * 128);
  float* dhf  = ws.f32((size_t)NTOK * 128);
  float* accum = ws.f32(16);
  int* sel     = ws.i32(NTOK);
  float* ysoft = ws.f32(NTOK * 8);
  float* phead = ws.f32(NTOK * 16);
  unsigned long long* best = ws.u64(8 * NTOK);
  float* gxr = a2tok;
  float* gd1 = a1;
  float* dt32 = tbuf;
  float* du32 = ubuf;
  float* dq32 = q32;
  float* gnv  = x0;
  float* dz0  = x1;

  hipMemsetAsync(accum, 0, 16 * sizeof(float), stream);
  hipMemsetAsync(best, 0xFF, (size_t)8 * NTOK * 8, stream);

  auto cvt = [&](const float* s, _Float16* d, int n) {
    k_cvt_f16<<<cdiv(n, 256), 256, 0, stream>>>(s, d, n);
  };
  auto gemm = [&](const _Float16* A, const _Float16* W, const float* bias, const float* resid,
                  float* C, _Float16* C16, int M, int N, int K, int flags) {
    int mt = cdiv(M, 16), nt = cdiv(N, 16);
    int gsz = ((nt & 3) == 0) ? 4 : (((nt & 1) == 0) ? 2 : 1);
    int ng  = (gsz == 4) ? (nt >> 2) : ((gsz == 2) ? (nt >> 1) : nt);
    int waves = mt * ng;
    k_wmma_gemm<<<cdiv(waves, 8), 256, 0, stream>>>(A, W, bias, resid, C, C16, M, N, K, flags);
  };
  auto im2col = [&](const float* T, int C, int Kp) {
    int n = NTOK * Kp;
    k_im2col3x3_tok<<<cdiv(n, 256), 256, 0, stream>>>(T, col, C, Kp);
  };
  auto relu = [&](const float* a, float* b, int n) {
    k_relu<<<cdiv(n, 256), 256, 0, stream>>>(a, b, n);
  };

  // ---- weight prep ----
  cvt(c3w, w_c3_16, 128 * 1152);
  for (int b = 0; b < 2; ++b) {
    cvt(ewa[b], ewa16[b], 32 * 1152);
    cvt(ewb[b], ewb16[b], 128 * 32);
    cvt(dwa[b], dwa16[b], 32 * 1152);
    cvt(dwb[b], dwb16[b], 128 * 32);
    k_transpose_pad_f16<<<cdiv(32 * 128, 256), 256, 0, stream>>>(dwb[b], dwbT16[b], 128, 32, 128);
    k_repack3x3_rot<<<cdiv(128 * 288, 256), 256, 0, stream>>>(dwa[b], dwarot16[b], 32, 128, 288);
  }
  cvt(wq, wq16, 128 * 128);
  cvt(wk, wk16, 128 * 128);
  cvt(qk, qk16, 8 * 128);
  k_transpose_pad_f16<<<cdiv(128 * 128, 256), 256, 0, stream>>>(wq, wqT16, 128, 128, 128);
  for (int i = 0; i < 8; ++i) {
    cvt(mcw[i], mcw16[i], DL[i] * 128);
    k_transpose_pad_f16<<<cdiv(128 * DP[i], 256), 256, 0, stream>>>(mcw[i], mcwT16[i], DL[i], 128, DP[i]);
    k_pad_rows_f16<<<cdiv(128 * K9P[i], 256), 256, 0, stream>>>(m1w[i], m1w16[i], 128, DL[i] * 9, K9P[i]);
    k_repack3x3_rot<<<cdiv(DL[i] * 1152, 256), 256, 0, stream>>>(m1w[i], m1rot16[i], 128, DL[i], 1152);
    k_pad_rows_f16<<<cdiv(KL[i] * DP[i], 256), 256, 0, stream>>>(emb[i], E16[i], KL[i], DL[i], DP[i]);
    k_enorm<<<cdiv(KL[i], 256), 256, 0, stream>>>(emb[i], enorm[i], KL[i], DL[i]);
  }

  // ================= encoder =================
  k_conv4x4s2<<<cdiv(64 * 64 * 256, 256), 256, 0, stream>>>(x, c1w, c1b, a1, 64, 3, 64, 32, 32,
                                                            CF_RELU | CF_BIAS);
  k_conv4x4s2<<<cdiv(64 * 128 * 64, 256), 256, 0, stream>>>(a1, c2w, c2b, a2tok, 64, 64, 128, 16, 16,
                                                            CF_RELU | CF_BIAS | CF_OUTTOK);
  im2col(a2tok, 128, 1152);
  gemm(col, w_c3_16, c3b, nullptr, x0, nullptr, NTOK, 128, 1152, GF_BIAS);
  relu(x0, ubuf, NTOK * 128);
  im2col(ubuf, 128, 1152);
  gemm(col, ewa16[0], nullptr, nullptr, tbuf, t16, NTOK, 32, 1152, GF_RELU | GF_F16OUT);
  gemm(t16, ewb16[0], nullptr, x0, x1, nullptr, NTOK, 128, 32, GF_RESID);
  relu(x1, ubuf, NTOK * 128);
  im2col(ubuf, 128, 1152);
  gemm(col, ewa16[1], nullptr, nullptr, tbuf, t16, NTOK, 32, 1152, GF_RELU | GF_F16OUT);
  gemm(t16, ewb16[1], nullptr, x1, x2, nullptr, NTOK, 128, 32, GF_RESID);
  relu(x2, htok, NTOK * 128);
  cvt(htok, h16, NTOK * 128);
  k_tok_to_nchw<<<cdiv(NTOK * 128, 256), 256, 0, stream>>>(htok, out + OFF_H, 64, 128, 64);

  // ================= attention =================
  gemm(h16, wq16, bq, nullptr, q32, nullptr, NTOK, 128, 128, GF_BIAS);
  gemm(qk16, wk16, bk, nullptr, kk32, nullptr, 8, 128, 128, GF_BIAS);
  k_attn_fwd<<<cdiv(NTOK, 128), 128, 0, stream>>>(q32, kk32, gum, sel, ysoft, phead, NTOK);

  // ================= VQ levels =================
  for (int i = 0; i < 8; ++i) {
    gemm(h16, mcw16[i], mcb[i], nullptr, Zf[i], nullptr, NTOK, DL[i], 128, GF_BIAS);
    k_pad_rows_f16<<<cdiv(NTOK * DP[i], 256), 256, 0, stream>>>(Zf[i], Z16[i], NTOK, DL[i], DP[i]);
    int chunks = KL[i] / VQ_ROWS;
    int gy = chunks < 32 ? chunks : 32;
    dim3 gr(NTOK / 16, gy);
    size_t shbytes = (size_t)VQ_ROWS * DP[i] * 2;
    k_vq_argmin<<<gr, 256, shbytes, stream>>>(Z16[i], E16[i], enorm[i], best + (size_t)i * NTOK,
                                              NTOK, KL[i], DP[i]);
    k_vq_finish<<<cdiv(NTOK * DL[i], 256), 256, 0, stream>>>(best + (size_t)i * NTOK, emb[i],
                                                             Zf[i], Qb[i], accum + 1 + i, NTOK, DL[i]);
    im2col(Qb[i], DL[i], K9P[i]);
    gemm(col, m1w16[i], m1b[i], nullptr, nv + (size_t)i * NTOK * 128, nullptr,
         NTOK, 128, K9P[i], GF_BIAS);
  }
  k_combine_state<<<cdiv(NTOK * 128, 256), 256, 0, stream>>>(sel, nv, state, NTOK);

  // ================= decoder =================
  relu(state, u0d, NTOK * 128);
  im2col(u0d, 128, 1152);
  gemm(col, dwa16[0], nullptr, nullptr, t0d, t16, NTOK, 32, 1152, GF_RELU | GF_F16OUT);
  gemm(t16, dwb16[0], nullptr, state, ds1, nullptr, NTOK, 128, 32, GF_RESID);
  relu(ds1, u1d, NTOK * 128);
  im2col(u1d, 128, 1152);
  gemm(col, dwa16[1], nullptr, nullptr, t1d, t16, NTOK, 32, 1152, GF_RELU | GF_F16OUT);
  gemm(t16, dwb16[1], nullptr, ds1, ds2, nullptr, NTOK, 128, 32, GF_RESID);
  relu(ds2, ytok, NTOK * 128);
  k_convT4x4s2<<<cdiv(64 * 64 * 256, 256), 256, 0, stream>>>(ytok, t1w, t1b, d1, 64, 128, 64, 8, 8,
                                                             CF_RELU | CF_INTOK);
  k_convT4x4s2<<<cdiv(64 * 3 * 1024, 256), 256, 0, stream>>>(d1, t2w, t2b, out + OFF_XR,
                                                             64, 64, 3, 16, 16, 0);
  k_sqdiff<<<cdiv(196608, 256), 256, 0, stream>>>(x, out + OFF_XR, accum, 196608);
  k_finalize<<<1, 1, 0, stream>>>(accum, out);

  // ================= backward: grad_h =================
  k_mse_grad<<<cdiv(196608, 256), 256, 0, stream>>>(x, out + OFF_XR, gxr, 196608, 2.f / 196608.f);
  k_conv4x4s2<<<cdiv(64 * 64 * 256, 256), 256, 0, stream>>>(gxr, t2w, nullptr, gd1,
                                                            64, 3, 64, 32, 32, 0);
  k_relu_bwd<<<cdiv(64 * 64 * 256, 256), 256, 0, stream>>>(gd1, d1, 64 * 64 * 256);
  k_conv4x4s2<<<cdiv(NTOK * 128, 256), 256, 0, stream>>>(gd1, t1w, nullptr, gs,
                                                         64, 64, 128, 16, 16, CF_OUTTOK);
  k_relu_bwd<<<cdiv(NTOK * 128, 256), 256, 0, stream>>>(gs, ytok, NTOK * 128);
  for (int b = 1; b >= 0; --b) {
    float* tact = (b == 1) ? t1d : t0d;
    float* uact = (b == 1) ? u1d : u0d;
    cvt(gs, g16, NTOK * 128);
    gemm(g16, dwbT16[b], nullptr, nullptr, dt32, nullptr, NTOK, 32, 128, 0);
    k_relu_bwd<<<cdiv(NTOK * 32, 256), 256, 0, stream>>>(dt32, tact, NTOK * 32);
    im2col(dt32, 32, 288);
    gemm(col, dwarot16[b], nullptr, nullptr, du32, nullptr, NTOK, 128, 288, 0);
    k_relu_bwd<<<cdiv(NTOK * 128, 256), 256, 0, stream>>>(du32, uact, NTOK * 128);
    k_add<<<cdiv(NTOK * 128, 256), 256, 0, stream>>>(gs, du32, NTOK * 128);
  }
  k_att_bwd<<<cdiv(NTOK, 128), 128, 0, stream>>>(gs, nv, kk32, ysoft, phead, dq32, NTOK);
  cvt(dq32, dq16, NTOK * 128);
  gemm(dq16, wqT16, nullptr, nullptr, dhf, nullptr, NTOK, 128, 128, 0);
  for (int i = 0; i < 8; ++i) {
    k_select_grad<<<cdiv(NTOK * 128, 256), 256, 0, stream>>>(sel, gs, gnv, i, NTOK);
    im2col(gnv, 128, 1152);
    gemm(col, m1rot16[i], nullptr, nullptr, dz0, nullptr, NTOK, DL[i], 1152, 0);
    float coef = 2.f * 0.25f / (8.f * 4096.f * (float)DL[i]);
    k_vq_grad_z<<<cdiv(NTOK * DL[i], 256), 256, 0, stream>>>(dz0, Zf[i], Qb[i], NTOK * DL[i], coef);
    k_pad_rows_f16<<<cdiv(NTOK * DP[i], 256), 256, 0, stream>>>(dz0, dzp16, NTOK, DL[i], DP[i]);
    gemm(dzp16, mcwT16[i], nullptr, dhf, dhf, nullptr, NTOK, 128, DP[i], GF_RESID);
  }
  k_tok_to_nchw<<<cdiv(NTOK * 128, 256), 256, 0, stream>>>(dhf, out + OFF_GH, 64, 128, 64);
}
